// CTM_partpad_dict_BN_82815559402201
// MI455X (gfx1250) — matrix operand order repaired and checked by reading, not compiled, not run on hardware
//
#include <hip/hip_runtime.h>
#include <math.h>

typedef __attribute__((ext_vector_type(16))) _Float16 v16h;
typedef __attribute__((ext_vector_type(8)))  _Float16 v8h;
typedef __attribute__((ext_vector_type(8)))  float    v8f;

#define EPS    1e-6f
#define BN_EPS 1e-5f

constexpr int B   = 64;
constexpr int N   = 4096;
constexpr int C   = 64;
constexpr int COUT= 128;
constexpr int H   = 64;
constexpr int W   = 64;
constexpr int H2  = 32;
constexpr int W2  = 32;
constexpr int NS  = H2 * W2;        // 1024
constexpr int KCV = 9 * C;          // 576 im2col depth

__device__ __forceinline__ int cell_of(float lx, float ly, int hh, int ww) {
    lx = fminf(fmaxf(lx, -1.f), 1.f) * 0.5f + 0.5f;
    ly = fminf(fmaxf(ly, -1.f), 1.f) * 0.5f + 0.5f;
    int cx = (int)fminf(fmaxf(rintf(lx * (float)(ww - 1)), 0.f), (float)(ww - 1));
    int cy = (int)fminf(fmaxf(rintf(ly * (float)(hh - 1)), 0.f), (float)(hh - 1));
    return cx + cy * ww;
}

// concat two 8-half vectors into a 16-half WMMA fragment
__device__ __forceinline__ v16h cat8(v8h lo, v8h hi) {
    return __builtin_shufflevector(lo, hi, 0,1,2,3,4,5,6,7,8,9,10,11,12,13,14,15);
}

// gfx1250 async global->LDS copy (ASYNCcnt-tracked), with wait helper
__device__ __forceinline__ void async_g2l_b64(unsigned lds_off, const void* gaddr) {
    asm volatile("global_load_async_to_lds_b64 %0, %1, off" :: "v"(lds_off), "v"(gaddr) : "memory");
}
__device__ __forceinline__ void wait_async0() {
#if __has_builtin(__builtin_amdgcn_s_wait_asynccnt)
    __builtin_amdgcn_s_wait_asynccnt(0);
#else
    asm volatile("s_wait_asynccnt 0x0" ::: "memory");
#endif
}

// ---------------- zero init (float4 stores) ----------------
__global__ void k_zero4(float4* p, long n4) {
    long i = (long)blockIdx.x * blockDim.x + threadIdx.x;
    if (i < n4) p[i] = make_float4(0.f, 0.f, 0.f, 0.f);
}

// ---------------- f32 -> f16 convert ----------------
__global__ void k_cvt_h(const float* __restrict__ in, _Float16* __restrict__ out, long n) {
    long i = (long)blockIdx.x * blockDim.x + threadIdx.x;
    if (i < n) out[i] = (_Float16)in[i];
}

// ---------------- f32 [k][co] -> f16 transposed [co][k] ----------------
__global__ void k_cvt_ht(const float* __restrict__ in, _Float16* __restrict__ out, int K, int CO) {
    long i = (long)blockIdx.x * blockDim.x + threadIdx.x;
    if (i >= (long)K * CO) return;
    int co = (int)(i / K), k = (int)(i - (long)co * K);
    out[i] = (_Float16)in[(long)k * CO + co];
}

// ---------------- token2map scatter: 16 thread-groups of 4ch per point ----------------
__global__ void k_scatter_map(const float* __restrict__ x, const float* __restrict__ loc,
                              const int* __restrict__ idx_agg,
                              float* __restrict__ s_map, float* __restrict__ cnt) {
    long gid = (long)blockIdx.x * blockDim.x + threadIdx.x;   // B*N*16
    long point = gid >> 4;
    int  cg    = (int)(gid & 15) * 4;
    if (point >= (long)B * N) return;
    int b = (int)(point / N);
    int t = idx_agg[point];
    int cell = cell_of(loc[2 * point], loc[2 * point + 1], H, W);
    const float4 f = *(const float4*)&x[((long)b * N + t) * C + cg];
    float* dst = &s_map[((long)b * H * W + cell) * C + cg];
    atomicAdd(dst + 0, f.x);
    atomicAdd(dst + 1, f.y);
    atomicAdd(dst + 2, f.z);
    atomicAdd(dst + 3, f.w);
    if (cg == 0) atomicAdd(&cnt[(long)b * H * W + cell], 1.f);
}

// ---------------- per-cell average -> f16 map ----------------
__global__ void k_map_avg(const float* __restrict__ s_map, const float* __restrict__ cnt,
                          _Float16* __restrict__ xmaph) {
    long i = (long)blockIdx.x * blockDim.x + threadIdx.x;     // B*H*W*C
    if (i >= (long)B * H * W * C) return;
    xmaph[i] = (_Float16)(s_map[i] / (cnt[i >> 6] + EPS));
}

// ---------------- 3x3 stride-2 conv via WMMA (im2col in LDS) ----------------
// block = 256 thr (8 waves). Tile: 16 output pixels x 128 out channels.
// wT: f16 weights transposed to [co][k] so B-fragments are contiguous.
__global__ void k_conv_wmma(const _Float16* __restrict__ xmap, const _Float16* __restrict__ wT,
                            const float* __restrict__ conv_b, float* __restrict__ y) {
    __shared__ __align__(16) _Float16 As[16 * KCV];           // 18 KB
    int blk = blockIdx.x;                        // B * 32 * 2
    int owt = blk & 1;
    int oh  = (blk >> 1) & 31;
    int b   = blk >> 6;
    int tid = threadIdx.x;
    // stage im2col patch: 16 px x 9 taps x 64 ch, in 8-half (16B) chunks
    for (int i = tid; i < 16 * 9 * 8; i += 256) {
        int p  = i / 72;
        int r  = i - p * 72;
        int kxy = r >> 3;
        int c8  = (r & 7) * 8;
        int ky = kxy / 3, kx = kxy - ky * 3;
        int ow = owt * 16 + p;
        int ih = 2 * oh + ky - 1, iw = 2 * ow + kx - 1;
        v8h v = {};
        if (ih >= 0 && ih < H && iw >= 0 && iw < W)
            v = *(const v8h*)&xmap[(((long)b * H + ih) * W + iw) * C + c8];
        *(v8h*)&As[p * KCV + kxy * 64 + c8] = v;
    }
    __syncthreads();

    int wv = tid >> 5, lane = tid & 31;
    int nloc = lane & 15, hb = lane >> 4;
    int co = wv * 16 + nloc;
    int m  = lane & 15;
    __builtin_prefetch(&wT[(long)co * KCV], 0, 1);            // pull weight row toward cache
    v8f acc = {};
    for (int kb = 0; kb < KCV / 32; ++kb) {
        int kbase = kb * 32;
        // A: 16 contiguous halves (k = 16*hb + s), matching B
        v8h alo = *(const v8h*)&As[m * KCV + kbase + hb * 16];
        v8h ahi = *(const v8h*)&As[m * KCV + kbase + hb * 16 + 8];
        // B: 16 contiguous halves in transposed weights
        v16h bm = *(const v16h*)&wT[(long)co * KCV + kbase + hb * 16];
        acc = __builtin_amdgcn_wmma_f32_16x16x32_f16(false, cat8(alo, ahi), false, bm,
                                                     (short)0, acc, false, false);
    }
    float bias = conv_b[co];
#pragma unroll
    for (int r = 0; r < 8; ++r) {
        int px = r + hb * 8;
        int ow = owt * 16 + px;
        y[(((long)b * H2 + oh) * W2 + ow) * COUT + co] = acc[r] + bias;
    }
}

// ---------------- map2token scatter (gather y at cell, atomic into per-token sums) ----------------
__global__ void k_scatter_tok(const float* __restrict__ y, const float* __restrict__ loc,
                              const int* __restrict__ idx_agg, const float* __restrict__ agg_w,
                              float* __restrict__ s_tok, float* __restrict__ norm_tok) {
    long gid = (long)blockIdx.x * blockDim.x + threadIdx.x;   // B*N*32
    long point = gid >> 5;
    int  cg    = (int)(gid & 31) * 4;
    if (point >= (long)B * N) return;
    int b = (int)(point / N);
    int t = idx_agg[point];
    float v = agg_w[point];
    int cell = cell_of(loc[2 * point], loc[2 * point + 1], H2, W2);
    const float4 f = *(const float4*)&y[((long)b * NS + cell) * COUT + cg];
    float* dst = &s_tok[((long)b * N + t) * COUT + cg];
    atomicAdd(dst + 0, f.x * v);
    atomicAdd(dst + 1, f.y * v);
    atomicAdd(dst + 2, f.z * v);
    atomicAdd(dst + 3, f.w * v);
    if (cg == 0) atomicAdd(&norm_tok[(long)b * N + t], v);
}

// ---------------- skip GEMM via WMMA, fused with segment-mean combine (in place on s_tok) ----------------
// swT: f16 skip weights transposed to [co][k]
__global__ void k_skip_wmma(const _Float16* __restrict__ xh, const _Float16* __restrict__ swT,
                            const float* __restrict__ norm_tok, float* __restrict__ xt) {
    __shared__ __align__(16) _Float16 As[16 * 64];            // 2 KB
    long row0 = (long)blockIdx.x * 16;
    int tid = threadIdx.x;
    // async global->LDS staging (ASYNCcnt path): 256 thr x 8B = 2 KB
    async_g2l_b64((unsigned)(size_t)&As[tid * 4], &xh[row0 * 64 + tid * 4]);
    __builtin_prefetch(&xt[row0 * COUT + tid * 8], 0, 1);     // epilogue rows
    wait_async0();
    __syncthreads();

    int wv = tid >> 5, lane = tid & 31;
    int nloc = lane & 15, hb = lane >> 4;
    int co = wv * 16 + nloc;
    int m  = lane & 15;
    v8f acc = {};
#pragma unroll
    for (int kb = 0; kb < 2; ++kb) {
        v8h alo = *(const v8h*)&As[m * 64 + kb * 32 + hb * 16];
        v8h ahi = *(const v8h*)&As[m * 64 + kb * 32 + hb * 16 + 8];
        v16h bm = *(const v16h*)&swT[(long)co * 64 + kb * 32 + hb * 16];
        acc = __builtin_amdgcn_wmma_f32_16x16x32_f16(false, cat8(alo, ahi), false, bm,
                                                     (short)0, acc, false, false);
    }
#pragma unroll
    for (int r = 0; r < 8; ++r) {
        long row = row0 + r + hb * 8;
        long o = row * COUT + co;
        xt[o] = xt[o] / (norm_tok[row] + EPS) + acc[r];
    }
}

// ---------------- BN stats: register -> LDS -> 128-address atomics ----------------
__global__ void k_bn_stats(const float* __restrict__ xt, float* __restrict__ bn_sum,
                           float* __restrict__ bn_sq) {
    __shared__ float ls[256], lq[256];
    int tid = threadIdx.x;
    int c = tid & 127, rp = tid >> 7;
    float s = 0.f, q = 0.f;
    for (long row = (long)blockIdx.x * 2 + rp; row < (long)B * N; row += (long)gridDim.x * 2) {
        float v = xt[row * COUT + c];
        s += v; q += v * v;
    }
    ls[tid] = s; lq[tid] = q;
    __syncthreads();
    if (tid < 128) {
        atomicAdd(&bn_sum[c], ls[tid] + ls[tid + 128]);
        atomicAdd(&bn_sq[c],  lq[tid] + lq[tid + 128]);
    }
}

__global__ void k_bn_final(const float* __restrict__ bn_sum, const float* __restrict__ bn_sq,
                           const float* __restrict__ gamma, const float* __restrict__ beta,
                           float* __restrict__ scale, float* __restrict__ shift) {
    int c = threadIdx.x;
    float inv = 1.f / ((float)B * (float)N);
    float mean = bn_sum[c] * inv;
    float var  = bn_sq[c] * inv - mean * mean;
    float sc   = rsqrtf(var + BN_EPS) * gamma[c];
    scale[c] = sc;
    shift[c] = beta[c] - mean * sc;
}

// ---------------- BN apply + conf head + exp + relu (wave per row) ----------------
__global__ void k_bn_conf(float* __restrict__ xt, const float* __restrict__ scale,
                          const float* __restrict__ shift, const float* __restrict__ conf_w,
                          const float* __restrict__ conf_b, float* __restrict__ x_act,
                          float* __restrict__ conf_out, float* __restrict__ wexp) {
    long row = (long)blockIdx.x * 8 + (threadIdx.x >> 5);
    int lane = threadIdx.x & 31;
    int c0 = lane * 4;
    long o = row * COUT + c0;
    float4 v  = *(float4*)&xt[o];
    float4 sc = *(const float4*)&scale[c0];
    float4 sh = *(const float4*)&shift[c0];
    v.x = v.x * sc.x + sh.x;  v.y = v.y * sc.y + sh.y;
    v.z = v.z * sc.z + sh.z;  v.w = v.w * sc.w + sh.w;
    float4 cw = *(const float4*)&conf_w[c0];
    float part = v.x * cw.x + v.y * cw.y + v.z * cw.z + v.w * cw.w;
#pragma unroll
    for (int off = 16; off > 0; off >>= 1) part += __shfl_xor(part, off, 32);
    float conf = part + conf_b[0];
    *(float4*)&xt[o] = v;                                   // normalized xt (in place)
    float4 r = make_float4(fmaxf(v.x, 0.f), fmaxf(v.y, 0.f), fmaxf(v.z, 0.f), fmaxf(v.w, 0.f));
    *(float4*)&x_act[o] = r;
    if (lane == 0) { conf_out[row] = conf; wexp[row] = expf(conf); }
}

// ---------------- token_cluster_grid scatter ----------------
__global__ void k_scatter_down(const float* __restrict__ xt, const float* __restrict__ loc,
                               const int* __restrict__ idx_agg, const float* __restrict__ wexp,
                               float* __restrict__ s_down, float* __restrict__ norm_down) {
    long gid = (long)blockIdx.x * blockDim.x + threadIdx.x;   // B*N*32
    long point = gid >> 5;
    int  cg    = (int)(gid & 31) * 4;
    if (point >= (long)B * N) return;
    int b = (int)(point / N);
    int t = idx_agg[point];
    float v = wexp[(long)b * N + t];
    int cell = cell_of(loc[2 * point], loc[2 * point + 1], H2, W2);
    const float4 f = *(const float4*)&xt[((long)b * N + t) * COUT + cg];
    float* dst = &s_down[((long)b * NS + cell) * COUT + cg];
    atomicAdd(dst + 0, f.x * v);
    atomicAdd(dst + 1, f.y * v);
    atomicAdd(dst + 2, f.z * v);
    atomicAdd(dst + 3, f.w * v);
    if (cg == 0) atomicAdd(&norm_down[(long)b * NS + cell], v);
}

// ---------------- x_down = relu(sum/norm) ----------------
__global__ void k_xdown(const float* __restrict__ s_down, const float* __restrict__ norm_down,
                        float* __restrict__ out) {
    long i = (long)blockIdx.x * blockDim.x + threadIdx.x;     // B*NS*COUT
    if (i >= (long)B * NS * COUT) return;
    out[i] = fmaxf(s_down[i] / (norm_down[i >> 7] + EPS), 0.f);
}

// ---------------- per-point weight_t, idx out, per-batch max ----------------
__global__ void k_point_final(const float* __restrict__ loc, const int* __restrict__ idx_agg,
                              const float* __restrict__ agg_w, const float* __restrict__ wexp,
                              const float* __restrict__ norm_down, float* __restrict__ awd,
                              float* __restrict__ idx_out, unsigned int* __restrict__ maxb) {
    long point = (long)blockIdx.x * blockDim.x + threadIdx.x;  // B*N
    if (point >= (long)B * N) return;
    int b = (int)(point / N);
    int t = idx_agg[point];
    int cell = cell_of(loc[2 * point], loc[2 * point + 1], H2, W2);
    float v  = wexp[(long)b * N + t];
    float wt = v / (norm_down[(long)b * NS + cell] + EPS);
    float a  = agg_w[point] * wt;                              // > 0 always
    awd[point] = a;
    idx_out[point] = (float)cell;
    atomicMax(&maxb[b], __float_as_uint(a));                   // positive floats: uint order == float order
}

__global__ void k_agg_norm(const float* __restrict__ awd, const unsigned int* __restrict__ maxb,
                           float* __restrict__ out) {
    long point = (long)blockIdx.x * blockDim.x + threadIdx.x;  // B*N
    if (point >= (long)B * N) return;
    int b = (int)(point / N);
    out[point] = awd[point] / __uint_as_float(maxb[b]);
}

extern "C" void kernel_launch(void* const* d_in, const int* in_sizes, int n_in,
                              void* d_out, int out_size, void* d_ws, size_t ws_size,
                              hipStream_t stream) {
    const float* x        = (const float*)d_in[0];
    const float* loc      = (const float*)d_in[1];
    const int*   idx_agg  = (const int*)  d_in[2];
    const float* agg_w    = (const float*)d_in[3];
    const float* conv_w   = (const float*)d_in[4];
    const float* conv_b   = (const float*)d_in[5];
    const float* skip_w   = (const float*)d_in[6];
    const float* bn_gamma = (const float*)d_in[7];
    const float* bn_beta  = (const float*)d_in[8];
    const float* conf_w   = (const float*)d_in[9];
    const float* conf_b   = (const float*)d_in[10];
    (void)in_sizes; (void)n_in; (void)out_size; (void)ws_size;

    // ---- workspace layout: zero-needed region first (contiguous) ----
    char* w = (char*)d_ws;
    float* s_map    = (float*)w;                     w += (size_t)B * H * W * C * 4;    // 64 MB
    float* cntm     = (float*)w;                     w += (size_t)B * H * W * 4;        // 1 MB
    float* s_tok    = (float*)w;                     w += (size_t)B * N * COUT * 4;     // 128 MB (becomes xt)
    float* norm_tok = (float*)w;                     w += (size_t)B * N * 4;            // 1 MB
    float* bn_sum   = (float*)w;                     w += 128 * 4;
    float* bn_sq    = (float*)w;                     w += 128 * 4;
    float* bn_scale = (float*)w;                     w += 128 * 4;
    float* bn_shift = (float*)w;                     w += 128 * 4;
    float* s_down   = (float*)w;                     w += (size_t)B * NS * COUT * 4;    // 32 MB
    float* norm_dn  = (float*)w;                     w += (size_t)B * NS * 4;           // 256 KB
    unsigned int* maxb = (unsigned int*)w;           w += 64 * 4;
    long zero_floats = (long)(((float*)w) - s_map);
    // non-zeroed scratch
    _Float16* xmaph = (_Float16*)w;                  w += (size_t)B * H * W * C * 2;    // 32 MB
    _Float16* wT    = (_Float16*)w;                  w += (size_t)KCV * COUT * 2;       // [co][k]
    _Float16* swT   = (_Float16*)w;                  w += (size_t)C * COUT * 2;         // [co][k]
    _Float16* xh    = (_Float16*)w;                  w += (size_t)B * N * C * 2;        // 32 MB
    float* y        = (float*)w;                     w += (size_t)B * NS * COUT * 4;    // 32 MB
    float* wexp     = (float*)w;                     w += (size_t)B * N * 4;            // 1 MB
    float* awd      = (float*)w;                     w += (size_t)B * N * 4;            // 1 MB

    float* out     = (float*)d_out;
    float* o_xdown = out;                                   // B*NS*COUT
    float* o_idx   = o_xdown + (long)B * NS * COUT;         // B*N
    float* o_aggw  = o_idx   + (long)B * N;                 // B*N
    float* o_xact  = o_aggw  + (long)B * N;                 // B*N*COUT
    float* o_conf  = o_xact  + (long)B * N * COUT;          // B*N

    const int T = 256;
    // 0) zero accumulation buffers
    long n4 = zero_floats / 4;
    k_zero4<<<(unsigned)((n4 + T - 1) / T), T, 0, stream>>>((float4*)s_map, n4);
    // 1) f16 converts (weights transposed to [co][k])
    k_cvt_ht<<<(KCV * COUT + T - 1) / T, T, 0, stream>>>(conv_w, wT, KCV, COUT);
    k_cvt_ht<<<(C * COUT + T - 1) / T, T, 0, stream>>>(skip_w, swT, C, COUT);
    k_cvt_h<<<(unsigned)(((long)B * N * C + T - 1) / T), T, 0, stream>>>(x, xh, (long)B * N * C);
    // 2) token2map scatter + average
    k_scatter_map<<<(unsigned)(((long)B * N * 16) / T), T, 0, stream>>>(x, loc, idx_agg, s_map, cntm);
    k_map_avg<<<(unsigned)(((long)B * H * W * C) / T), T, 0, stream>>>(s_map, cntm, xmaph);
    // 3) conv via WMMA
    k_conv_wmma<<<B * 32 * 2, T, 0, stream>>>(xmaph, wT, conv_b, y);
    // 4) map2token scatter, then skip WMMA + combine (in-place -> xt lives in s_tok)
    k_scatter_tok<<<(unsigned)(((long)B * N * 32) / T), T, 0, stream>>>(y, loc, idx_agg, agg_w, s_tok, norm_tok);
    k_skip_wmma<<<(unsigned)((long)B * N / 16), T, 0, stream>>>(xh, swT, norm_tok, s_tok);
    // 5) BatchNorm
    k_bn_stats<<<1024, T, 0, stream>>>(s_tok, bn_sum, bn_sq);
    k_bn_final<<<1, 128, 0, stream>>>(bn_sum, bn_sq, bn_gamma, bn_beta, bn_scale, bn_shift);
    k_bn_conf<<<(unsigned)((long)B * N / 8), T, 0, stream>>>(s_tok, bn_scale, bn_shift, conf_w, conf_b,
                                                             o_xact, o_conf, wexp);
    // 6) token_cluster_grid
    k_scatter_down<<<(unsigned)(((long)B * N * 32) / T), T, 0, stream>>>(s_tok, loc, idx_agg, wexp, s_down, norm_dn);
    k_xdown<<<(unsigned)(((long)B * NS * COUT) / T), T, 0, stream>>>(s_down, norm_dn, o_xdown);
    k_point_final<<<(unsigned)((long)B * N / T), T, 0, stream>>>(loc, idx_agg, agg_w, wexp, norm_dn,
                                                                 awd, o_idx, maxb);
    k_agg_norm<<<(unsigned)((long)B * N / T), T, 0, stream>>>(awd, maxb, o_aggw);
}
